// MicroStreamBlock_19731079758488
// MI455X (gfx1250) — compile-verified, matrix operand order checked by reading, not run on hardware
//
#include <hip/hip_runtime.h>
#include <math.h>

// ---------------------------------------------------------------------------
// MicroStreamBlock for MI455X (gfx1250, wave32, WMMA).
// GEMMs (fused QKV 8192x3072x1024 + two FFN 8192x1024x1024) run on
// v_wmma_f32_16x16x32_bf16 with double-buffered LDS fed by
// global_load_async_to_lds_b128 (ASYNCcnt). KTILE=64: 16 WMMAs per barrier.
// ---------------------------------------------------------------------------

typedef __attribute__((ext_vector_type(16))) __bf16 v16bf;
typedef __attribute__((ext_vector_type(8)))  float  v8f;

union ABFrag { v16bf v; uint4 u[2]; };

#define DIMD 1024
#define NQKV 3072
#define MROWS 8192        // B*T
#define MT 128
#define NT 128
#define KTILE 64          // staged K per LDS tile (two 32-wide WMMA k-halves)

// one 16-byte async copy global->LDS (ASYNCcnt-tracked)
#define ASYNC_CP16(ldsoff, gptr)                                              \
    asm volatile("global_load_async_to_lds_b128 %0, %1, off"                  \
                 :: "v"(ldsoff), "v"((unsigned long long)(gptr)) : "memory")

__device__ __forceinline__ unsigned lds_off32(const void* p) {
    return (unsigned)(unsigned long long)p;   // low 32 bits = LDS byte offset
}

// ---------------------------------------------------------------------------
// Weight prep:
//  - wqkvb[n][k] (n in [0,3072)): bf16 of [wq; wk; wv] row-major (x @ w.T layout)
//  - w1t/w2t[n][k] = (1/sqrt(8)) * sum_r f[r][k][n]   (transposed rank-sum)
//  - bcat = [bq; bk; bv]
// ---------------------------------------------------------------------------
__global__ __launch_bounds__(256)
void prep_weights(const float* __restrict__ wq, const float* __restrict__ wk,
                  const float* __restrict__ wv,
                  const float* __restrict__ bq, const float* __restrict__ bk,
                  const float* __restrict__ bv,
                  const float* __restrict__ f1, const float* __restrict__ f2,
                  __bf16* __restrict__ wqkvb,
                  __bf16* __restrict__ w1t, __bf16* __restrict__ w2t,
                  float* __restrict__ bcat)
{
    size_t idx = (size_t)blockIdx.x * 256 + threadIdx.x;
    if (idx >= (size_t)NQKV * DIMD) return;
    int n = (int)(idx >> 10);
    int k = (int)(idx & 1023);

    float wsrc;
    if (n < 1024)      wsrc = wq[(size_t)n * DIMD + k];
    else if (n < 2048) wsrc = wk[(size_t)(n - 1024) * DIMD + k];
    else               wsrc = wv[(size_t)(n - 2048) * DIMD + k];
    wqkvb[idx] = (__bf16)wsrc;

    if (idx < (size_t)DIMD * DIMD) {
        size_t t = (size_t)k * DIMD + n;   // transposed read
        float s1 = 0.f, s2 = 0.f;
#pragma unroll
        for (int r = 0; r < 8; ++r) {
            s1 += f1[(size_t)r * DIMD * DIMD + t];
            s2 += f2[(size_t)r * DIMD * DIMD + t];
        }
        const float sc = 0.35355339059327373f;  // 1/sqrt(RANK)
        w1t[idx] = (__bf16)(s1 * sc);
        w2t[idx] = (__bf16)(s2 * sc);
    }
    if (idx < NQKV) {
        int j = (int)idx;
        bcat[j] = (j < 1024) ? bq[j] : (j < 2048) ? bk[j - 1024] : bv[j - 2048];
    }
}

// ---------------------------------------------------------------------------
// LayerNorm + bf16 cast. grid = (8192, 2): y==0 -> x2 with (n1_g,n1_b) -> ln2
//                                          y==1 -> x1 with (n2_g,n2_b) -> ln1
// ---------------------------------------------------------------------------
__global__ __launch_bounds__(256)
void layernorm_bf16(const float* __restrict__ x,
                    const float* __restrict__ g1, const float* __restrict__ be1,
                    const float* __restrict__ g2, const float* __restrict__ be2,
                    __bf16* __restrict__ ln2, __bf16* __restrict__ ln1)
{
    __shared__ float sred[256];
    const int row   = blockIdx.x;
    const int which = blockIdx.y;
    const float* src = x + (size_t)row * 2048 + (which == 0 ? 1024 : 0);
    const float* g   = (which == 0) ? g1 : g2;
    const float* be  = (which == 0) ? be1 : be2;
    __bf16* dst = ((which == 0) ? ln2 : ln1) + (size_t)row * DIMD;
    const int tid = threadIdx.x;

    float4 vv = reinterpret_cast<const float4*>(src)[tid];
    sred[tid] = vv.x + vv.y + vv.z + vv.w;
    __syncthreads();
    for (int st = 128; st > 0; st >>= 1) {
        if (tid < st) sred[tid] += sred[tid + st];
        __syncthreads();
    }
    float mean = sred[0] * (1.0f / 1024.0f);
    __syncthreads();
    float dx = vv.x - mean, dy = vv.y - mean, dz = vv.z - mean, dw = vv.w - mean;
    sred[tid] = dx * dx + dy * dy + dz * dz + dw * dw;
    __syncthreads();
    for (int st = 128; st > 0; st >>= 1) {
        if (tid < st) sred[tid] += sred[tid + st];
        __syncthreads();
    }
    float rstd = rsqrtf(sred[0] * (1.0f / 1024.0f) + 1e-5f);
    int i = tid * 4;
    dst[i + 0] = (__bf16)(dx * rstd * g[i + 0] + be[i + 0]);
    dst[i + 1] = (__bf16)(dy * rstd * g[i + 1] + be[i + 1]);
    dst[i + 2] = (__bf16)(dz * rstd * g[i + 2] + be[i + 2]);
    dst[i + 3] = (__bf16)(dw * rstd * g[i + 3] + be[i + 3]);
}

// ---------------------------------------------------------------------------
// WMMA GEMM:  C[M,N] = A[M,K] (bf16) x Bw[N,K] (bf16, B[k][n] = Bw[n*K+k]) + bias
//   MODE 0: store f32 to Cf (ld = N)
//   MODE 1: exact GELU, store bf16 to Cb
//   MODE 2: += residual xin[m*2048+1024+n], store f32 to yout (same indexing)
// Block = 256 threads (8 waves). Block tile 128x128, wave tile 16x128,
// staged K tile 64 (two 32-wide WMMA passes per barrier), double-buffered LDS
// filled by async global->LDS copies.
// ---------------------------------------------------------------------------
template <int MODE>
__global__ __launch_bounds__(256)
void gemm_bf16(const __bf16* __restrict__ A, const __bf16* __restrict__ Bw,
               const float* __restrict__ bias,
               float* __restrict__ Cf, __bf16* __restrict__ Cb,
               const float* __restrict__ xin, float* __restrict__ yout,
               int M, int N, int K)
{
    __shared__ __align__(16) __bf16 sA[2][MT * KTILE];   // 2 x 16 KB
    __shared__ __align__(16) __bf16 sB[2][NT * KTILE];   // 2 x 16 KB

    const int tid  = threadIdx.x;
    const int wave = tid >> 5;
    const int lane = tid & 31;
    const int m0 = blockIdx.y * MT;
    const int n0 = blockIdx.x * NT;
    const int row  = tid >> 1;     // 0..127 staging row
    const int half = tid & 1;      // 32-element half of the 64-wide K strip

    const int ml = wave * 16 + (lane & 15);   // A fragment row (M)
    const int kb = (lane >> 4) * 8;           // K octet base per lane group

    v8f acc[8] = {};

    const __bf16* gA = A  + (size_t)(m0 + row) * K + half * 32;
    const __bf16* gB = Bw + (size_t)(n0 + row) * K + half * 32;
    const unsigned stA = lds_off32(&sA[0][row * KTILE + half * 32]);
    const unsigned stB = lds_off32(&sB[0][row * KTILE + half * 32]);
    const unsigned bufbytes = MT * KTILE * sizeof(__bf16);   // 16384

    // prologue: async-stage tile 0 into buffer 0 (64 B per matrix per thread)
#pragma unroll
    for (int c = 0; c < 4; ++c) {
        ASYNC_CP16(stA + c * 16, gA + c * 8);
        ASYNC_CP16(stB + c * 16, gB + c * 8);
    }
    asm volatile("s_wait_asynccnt 0" ::: "memory");
    __syncthreads();

    int cur = 0;
    for (int k0 = 0; k0 < K; k0 += KTILE) {
        const bool has_next = (k0 + KTILE) < K;
        if (has_next) {
            const int kn = k0 + KTILE;
            const unsigned nb = (cur ^ 1) * bufbytes;
#pragma unroll
            for (int c = 0; c < 4; ++c) {
                ASYNC_CP16(stA + nb + c * 16, gA + kn + c * 8);
                ASYNC_CP16(stB + nb + c * 16, gB + kn + c * 8);
            }
            if (kn + KTILE < K) {   // warm GL2 for tile k+2
                __builtin_prefetch(gA + kn + KTILE, 0, 0);
                __builtin_prefetch(gB + kn + KTILE, 0, 0);
            }
        }

#pragma unroll
        for (int kh = 0; kh < 2; ++kh) {   // two 32-wide WMMA passes
            ABFrag af;
            const uint4* ap = reinterpret_cast<const uint4*>(
                &sA[cur][ml * KTILE + kh * 32 + kb]);
            af.u[0] = ap[0];      // K[kb .. kb+8)
            af.u[1] = ap[2];      // K[kb+16 .. kb+24)
#pragma unroll
            for (int g = 0; g < 2; ++g) {
                ABFrag bf[4];
#pragma unroll
                for (int i = 0; i < 4; ++i) {
                    const uint4* bp = reinterpret_cast<const uint4*>(
                        &sB[cur][((g * 4 + i) * 16 + (lane & 15)) * KTILE +
                                 kh * 32 + kb]);
                    bf[i].u[0] = bp[0];
                    bf[i].u[1] = bp[2];
                }
#pragma unroll
                for (int i = 0; i < 4; ++i) {
                    acc[g * 4 + i] = __builtin_amdgcn_wmma_f32_16x16x32_bf16(
                        false, af.v, false, bf[i].v, (short)0, acc[g * 4 + i],
                        false, false);
                }
            }
        }

        if (has_next) asm volatile("s_wait_asynccnt 0" ::: "memory");
        __syncthreads();
        cur ^= 1;
    }

    const int lm = (lane >> 4) * 8;
    const int ln = lane & 15;
#pragma unroll
    for (int nf = 0; nf < 8; ++nf) {
        int n = n0 + nf * 16 + ln;
        float bv = bias[n];
#pragma unroll
        for (int j = 0; j < 8; ++j) {
            int m = m0 + wave * 16 + lm + j;
            float v = acc[nf][j] + bv;
            if (MODE == 0) {
                Cf[(size_t)m * N + n] = v;
            } else if (MODE == 1) {
                float g = 0.5f * v * (1.0f + erff(v * 0.70710678118654752f));
                Cb[(size_t)m * N + n] = (__bf16)g;
            } else {
                size_t o = (size_t)m * 2048 + 1024 + n;
                yout[o] = v + xin[o];
            }
        }
    }
}

// ---------------------------------------------------------------------------
// Quaternion logits from the fused qkv buffer (row stride 3072, q at +0,
// k at +1024). Stored transposed: logits[(b*256+hq)*2048 + t].
// ---------------------------------------------------------------------------
__global__ __launch_bounds__(256)
void quat_logits(const float* __restrict__ qkv, float* __restrict__ logits)
{
    const int row = blockIdx.x;       // b*2048 + t
    const int hq  = threadIdx.x;      // 0..255
    const size_t base = (size_t)row * NQKV + hq * 4;
    float4 qv = *reinterpret_cast<const float4*>(qkv + base);
    float4 kv = *reinterpret_cast<const float4*>(qkv + base + 1024);
    float qn2 = qv.x * qv.x + qv.y * qv.y + qv.z * qv.z + qv.w * qv.w;
    float kn2 = kv.x * kv.x + kv.y * kv.y + kv.z * kv.z + kv.w * kv.w;
    float rq = rsqrtf(qn2 + 1e-12f);
    float rk = rsqrtf(kn2 + 1e-12f);
    float dot = (qv.x * kv.x + qv.y * kv.y + qv.z * kv.z + qv.w * kv.w) * rq * rk;
    int b = row >> 11, t = row & 2047;
    logits[((size_t)(b * 256 + hq)) * 2048 + t] = dot * 0.125f;  // / sqrt(64)
}

// ---------------------------------------------------------------------------
// Softmax over t (2048) + weighted pooling of v (at qkv col offset 2048).
// ---------------------------------------------------------------------------
__global__ __launch_bounds__(256)
void attn_pool(const float* __restrict__ logits, const float* __restrict__ qkv,
               float* __restrict__ pooled)
{
    __shared__ float  sred[256];
    __shared__ float4 sacc[256];
    const int bh = blockIdx.x;        // b*256 + hq
    const int b = bh >> 8, hq = bh & 255;
    const int tid = threadIdx.x;
    const float* lrow = logits + (size_t)bh * 2048;

    float m = -3.0e38f;
    for (int t = tid; t < 2048; t += 256) m = fmaxf(m, lrow[t]);
    sred[tid] = m;
    __syncthreads();
    for (int st = 128; st > 0; st >>= 1) {
        if (tid < st) sred[tid] = fmaxf(sred[tid], sred[tid + st]);
        __syncthreads();
    }
    float M = sred[0];
    __syncthreads();

    float s = 0.f;
    float4 a = {0.f, 0.f, 0.f, 0.f};
    for (int t = tid; t < 2048; t += 256) {
        float p = expf(lrow[t] - M);
        s += p;
        const float* vp = qkv + ((size_t)(b * 2048 + t)) * NQKV + 2048 + hq * 4;
        a.x += p * vp[0]; a.y += p * vp[1]; a.z += p * vp[2]; a.w += p * vp[3];
    }
    sred[tid] = s;
    sacc[tid] = a;
    __syncthreads();
    for (int st = 128; st > 0; st >>= 1) {
        if (tid < st) {
            sred[tid] += sred[tid + st];
            sacc[tid].x += sacc[tid + st].x;
            sacc[tid].y += sacc[tid + st].y;
            sacc[tid].z += sacc[tid + st].z;
            sacc[tid].w += sacc[tid + st].w;
        }
        __syncthreads();
    }
    if (tid == 0) {
        float inv = 1.0f / sred[0];
        float* o = pooled + (size_t)b * DIMD + hq * 4;
        o[0] = sacc[0].x * inv;
        o[1] = sacc[0].y * inv;
        o[2] = sacc[0].z * inv;
        o[3] = sacc[0].w * inv;
    }
}

// ---------------------------------------------------------------------------
// Output projection of the pooled vector (tiny: 4x1024 outputs).
// ---------------------------------------------------------------------------
__global__ __launch_bounds__(256)
void attn_proj(const float* __restrict__ pooled, const float* __restrict__ wo,
               const float* __restrict__ bo, float* __restrict__ attnout)
{
    __shared__ float sred[256];
    const int b = blockIdx.x >> 10, j = blockIdx.x & 1023;
    const int tid = threadIdx.x;
    const float* pp = pooled + (size_t)b * DIMD;
    const float* wp = wo + (size_t)j * DIMD;
    float s = 0.f;
    for (int i = tid; i < DIMD; i += 256) s += pp[i] * wp[i];
    sred[tid] = s;
    __syncthreads();
    for (int st = 128; st > 0; st >>= 1) {
        if (tid < st) sred[tid] += sred[tid + st];
        __syncthreads();
    }
    if (tid == 0) attnout[(size_t)b * DIMD + j] = sred[0] + bo[j];
}

// ---------------------------------------------------------------------------
// y1 = x1 + attnout (broadcast over t); writes out[..., 0:1024].
// ---------------------------------------------------------------------------
__global__ __launch_bounds__(256)
void add_attn_y1(const float* __restrict__ x, const float* __restrict__ attnout,
                 float* __restrict__ out)
{
    const int row = blockIdx.x;           // b*2048 + t
    const int tid = threadIdx.x;
    const int b = row >> 11;
    float4 xv = reinterpret_cast<const float4*>(x + (size_t)row * 2048)[tid];
    float4 av = reinterpret_cast<const float4*>(attnout + (size_t)b * DIMD)[tid];
    float4 r = {xv.x + av.x, xv.y + av.y, xv.z + av.z, xv.w + av.w};
    reinterpret_cast<float4*>(out + (size_t)row * 2048)[tid] = r;
}

// ---------------------------------------------------------------------------
// Launcher
// ---------------------------------------------------------------------------
extern "C" void kernel_launch(void* const* d_in, const int* in_sizes, int n_in,
                              void* d_out, int out_size, void* d_ws, size_t ws_size,
                              hipStream_t stream)
{
    const float* x   = (const float*)d_in[0];
    const float* n1g = (const float*)d_in[1];
    const float* n1b = (const float*)d_in[2];
    const float* wq  = (const float*)d_in[3];
    const float* bq  = (const float*)d_in[4];
    const float* wk  = (const float*)d_in[5];
    const float* bk  = (const float*)d_in[6];
    const float* wv  = (const float*)d_in[7];
    const float* bv  = (const float*)d_in[8];
    const float* wo  = (const float*)d_in[9];
    const float* bo  = (const float*)d_in[10];
    const float* n2g = (const float*)d_in[11];
    const float* n2b = (const float*)d_in[12];
    const float* f1  = (const float*)d_in[13];
    const float* b1  = (const float*)d_in[14];
    const float* f2  = (const float*)d_in[15];
    const float* b2  = (const float*)d_in[16];
    float* out = (float*)d_out;
    char* ws = (char*)d_ws;

    const size_t MB = 1u << 20;
    __bf16* wqkvb = (__bf16*)(ws + 0 * MB);    // 6 MB  [3072][1024]
    __bf16* w1t   = (__bf16*)(ws + 6 * MB);    // 2 MB
    __bf16* w2t   = (__bf16*)(ws + 8 * MB);    // 2 MB
    float*  bcat  = (float*)(ws + 10 * MB);    // 12 KB
    __bf16* ln2   = (__bf16*)(ws + 11 * MB);   // 16 MB
    __bf16* ln1   = (__bf16*)(ws + 27 * MB);   // 16 MB
    float*  qkv   = (float*)(ws + 43 * MB);    // 96 MB [8192][3072]
    float*  logi  = (float*)(ws + 139 * MB);   // 8 MB
    float*  pool  = (float*)(ws + 147 * MB);   // 16 KB
    float*  atto  = (float*)(ws + 147 * MB + 65536);
    __bf16* h1    = (__bf16*)qkv;              // reuse qkv buffer after pooling

    // 1. weights (bf16 casts, fused qkv, transposed rank-summed FFN mats)
    prep_weights<<<(NQKV * DIMD + 255) / 256, 256, 0, stream>>>(
        wq, wk, wv, bq, bk, bv, f1, f2, wqkvb, w1t, w2t, bcat);
    // 2. layernorms (both halves)
    layernorm_bf16<<<dim3(MROWS, 2), 256, 0, stream>>>(x, n1g, n1b, n2g, n2b, ln2, ln1);
    // 3. fused QKV projection: [8192,1024] x [1024,3072]
    gemm_bf16<0><<<dim3(NQKV / NT, MROWS / MT), 256, 0, stream>>>(
        ln2, wqkvb, bcat, qkv, nullptr, nullptr, nullptr, MROWS, NQKV, DIMD);
    // 4. quaternion cosine logits
    quat_logits<<<MROWS, 256, 0, stream>>>(qkv, logi);
    // 5. softmax over t + v pooling
    attn_pool<<<4 * 256, 256, 0, stream>>>(logi, qkv, pool);
    // 6. wo projection (tiny)
    attn_proj<<<4 * 1024, 256, 0, stream>>>(pool, wo, bo, atto);
    // 7. y1 = x1 + broadcast attn output -> out[..., 0:1024]
    add_attn_y1<<<MROWS, 256, 0, stream>>>(x, atto, out);
    // 8. FFN1: ln1 @ W1 + b1 -> GELU -> bf16 (reuses qkv buffer)
    gemm_bf16<1><<<dim3(DIMD / NT, MROWS / MT), 256, 0, stream>>>(
        ln1, w1t, b1, nullptr, h1, nullptr, nullptr, MROWS, DIMD, DIMD);
    // 9. FFN2: h1 @ W2 + b2 + x2 -> out[..., 1024:2048]
    gemm_bf16<2><<<dim3(DIMD / NT, MROWS / MT), 256, 0, stream>>>(
        h1, w2t, b2, nullptr, nullptr, x, out, MROWS, DIMD, DIMD);
}
